// DynamicGraphConvolution_11046655886020
// MI455X (gfx1250) — matrix operand order repaired and checked by reading, not compiled, not run on hardware
//
#include <hip/hip_runtime.h>
#include <hip/hip_bf16.h>
#include <math.h>

typedef __bf16 bf16_t;
typedef __attribute__((ext_vector_type(16))) __bf16 v16bf;
typedef __attribute__((ext_vector_type(8)))  __bf16 v8bf;
typedef __attribute__((ext_vector_type(8)))  float  v8f;

#define BATCH 64
#define SEQ   1024
#define DIMD  256
#define QKV3  768

union AF { v16bf v; v8bf h[2]; };

// A fragment: 16 rows (m = lane&15) x 32 K. Per ISA 16-bit A layout:
// lanes 0-15: VGPR0-3 K=0..7, VGPR4-7 K=16..23 ; lanes 16-31: K=8..15 / 24..31
__device__ __forceinline__ v16bf load_a_frag(const bf16_t* __restrict__ Abase, int lda, int k0) {
  const int lane = threadIdx.x & 31;
  const int half = lane >> 4, r = lane & 15;
  AF f;
  const bf16_t* p = Abase + (size_t)r * lda + k0 + 8 * half;
  f.h[0] = *(const v8bf*)(p);        // K = k0 + 8h .. +7
  f.h[1] = *(const v8bf*)(p + 16);   // K = k0 + 16 + 8h .. +7
  return f.v;
}

// B fragment: B stored N-major rows of length K (NT gemm). n = lane&15,
// lanes 0-15 hold K = k0..k0+7 / k0+16..k0+23, lanes 16-31 hold K = k0+8..k0+15 / k0+24..k0+31.
__device__ __forceinline__ v16bf load_b_frag(const bf16_t* __restrict__ Bbase, int ldb, int k0) {
  const int lane = threadIdx.x & 31;
  const int half = lane >> 4, n = lane & 15;
  AF f;
  const bf16_t* p = Bbase + (size_t)n * ldb + k0 + 8 * half;
  f.h[0] = *(const v8bf*)(p);
  f.h[1] = *(const v8bf*)(p + 16);
  return f.v;
}

__device__ __forceinline__ v8f wmma_bf16(v16bf a, v16bf b, v8f c) {
  return __builtin_amdgcn_wmma_f32_16x16x32_bf16(false, a, false, b, (short)0, c, false, false);
}

__device__ __forceinline__ float leaky(float v) { return v > 0.f ? v : 0.2f * v; }

// ---------------- conversion / transpose ----------------
__global__ void k_cvt(const float* __restrict__ s, bf16_t* __restrict__ d, int n) {
  for (int i = blockIdx.x * blockDim.x + threadIdx.x; i < n; i += gridDim.x * blockDim.x)
    d[i] = (bf16_t)s[i];
}

__global__ void k_xT(const float* __restrict__ x, bf16_t* __restrict__ xT) {
  int i = blockIdx.x * blockDim.x + threadIdx.x;
  const int total = BATCH * DIMD * SEQ;
  if (i < total) {
    int n = i % SEQ; int t = i / SEQ; int d = t % DIMD; int b = t / DIMD;
    xT[i] = (bf16_t)x[((size_t)b * SEQ + n) * DIMD + d];
  }
}

// ---------------- K1: qkv = x @ qkv_w^T + b ; split q,k and v transposed ----------------
__global__ __launch_bounds__(256) void k_qkv(const bf16_t* __restrict__ xbf,
                                             const bf16_t* __restrict__ wbf,
                                             const float* __restrict__ bias,
                                             bf16_t* __restrict__ qb, bf16_t* __restrict__ kb,
                                             bf16_t* __restrict__ vT) {
  const int b = blockIdx.y;
  const int wave = threadIdx.x >> 5;
  const int tile = blockIdx.x * 8 + wave;         // 32 (M) x 24 (N) tiles
  const int m0 = (tile / 24) * 32, n0 = (tile % 24) * 32;
  const bf16_t* A = xbf + (size_t)b * SEQ * DIMD;
  v8f acc[2][2] = {};
  for (int k0 = 0; k0 < DIMD; k0 += 32) {
    v16bf a0 = load_a_frag(A + (size_t)m0 * DIMD, DIMD, k0);
    v16bf a1 = load_a_frag(A + (size_t)(m0 + 16) * DIMD, DIMD, k0);
    v16bf b0 = load_b_frag(wbf + (size_t)n0 * DIMD, DIMD, k0);
    v16bf b1 = load_b_frag(wbf + (size_t)(n0 + 16) * DIMD, DIMD, k0);
    acc[0][0] = wmma_bf16(a0, b0, acc[0][0]);
    acc[0][1] = wmma_bf16(a0, b1, acc[0][1]);
    acc[1][0] = wmma_bf16(a1, b0, acc[1][0]);
    acc[1][1] = wmma_bf16(a1, b1, acc[1][1]);
  }
  const int lane = threadIdx.x & 31, half = lane >> 4, nn = lane & 15;
  for (int i = 0; i < 2; i++)
    for (int j = 0; j < 2; j++) {
      int col = n0 + 16 * j + nn;
      float bi = bias[col];
      for (int r = 0; r < 8; r++) {
        int row = m0 + 16 * i + r + 8 * half;
        bf16_t bv = (bf16_t)(acc[i][j][r] + bi);
        if (col < 256)      qb[(size_t)b * SEQ * DIMD + (size_t)row * DIMD + col] = bv;
        else if (col < 512) kb[(size_t)b * SEQ * DIMD + (size_t)row * DIMD + (col - 256)] = bv;
        else                vT[(size_t)b * DIMD * SEQ + (size_t)(col - 512) * SEQ + row] = bv;
      }
    }
}

// ---------------- K2: flash attention + residual, writes x_glb transposed ----------------
__global__ __launch_bounds__(256) void k_attn(const bf16_t* __restrict__ qb,
                                              const bf16_t* __restrict__ kb,
                                              const bf16_t* __restrict__ vT,
                                              const float* __restrict__ x,
                                              const float* __restrict__ gamma,
                                              bf16_t* __restrict__ xglbT) {
  __shared__ bf16_t plds[8][16][32];
  const int b = blockIdx.y;
  const int wave = threadIdx.x >> 5;
  const int q0 = blockIdx.x * 128 + wave * 16;
  const int lane = threadIdx.x & 31, half = lane >> 4, nn = lane & 15;
  const bf16_t* Q = qb + (size_t)b * SEQ * DIMD + (size_t)q0 * DIMD;
  const bf16_t* K = kb + (size_t)b * SEQ * DIMD;
  const bf16_t* V = vT + (size_t)b * DIMD * SEQ;

  v16bf qa[8];
  for (int kk = 0; kk < 8; kk++) qa[kk] = load_a_frag(Q, DIMD, kk * 32);

  v8f acc[16] = {};                 // 16 query rows x full D=256 (16 col-tiles)
  float mrow[8], lrow[8];
  for (int r = 0; r < 8; r++) { mrow[r] = -3.0e38f; lrow[r] = 0.f; }

  for (int mc = 0; mc < SEQ; mc += 32) {
    v8f e0 = {}, e1 = {};
    for (int kk = 0; kk < 8; kk++) {
      v16bf b0 = load_b_frag(K + (size_t)mc * DIMD, DIMD, kk * 32);
      v16bf b1 = load_b_frag(K + (size_t)(mc + 16) * DIMD, DIMD, kk * 32);
      e0 = wmma_bf16(qa[kk], b0, e0);
      e1 = wmma_bf16(qa[kk], b1, e1);
    }
    // online softmax over this 32-column chunk; row r lives in 16 lanes of each half
    for (int r = 0; r < 8; r++) {
      float cm = fmaxf(e0[r], e1[r]);
      for (int off = 1; off < 16; off <<= 1) cm = fmaxf(cm, __shfl_xor(cm, off, 32));
      float mnew = fmaxf(mrow[r], cm);
      float scale = __expf(mrow[r] - mnew);
      float p0 = __expf(e0[r] - mnew);
      float p1 = __expf(e1[r] - mnew);
      float ps = p0 + p1;
      for (int off = 1; off < 16; off <<= 1) ps += __shfl_xor(ps, off, 32);
      lrow[r] = lrow[r] * scale + ps;
      mrow[r] = mnew;
      for (int t = 0; t < 16; t++) acc[t][r] *= scale;
      plds[wave][r + 8 * half][nn]      = (bf16_t)p0;   // C layout -> row-major LDS tile
      plds[wave][r + 8 * half][16 + nn] = (bf16_t)p1;
    }
    asm volatile("" ::: "memory");  // DS ops are in-order within a wave
    AF pf;                           // re-read as A-fragment (rows = query, K = 32 keys)
    pf.h[0] = *(const v8bf*)&plds[wave][nn][8 * half];
    pf.h[1] = *(const v8bf*)&plds[wave][nn][16 + 8 * half];
    asm volatile("" ::: "memory");
    for (int t = 0; t < 16; t++) {
      v16bf bv = load_b_frag(V + (size_t)(t * 16) * SEQ, SEQ, mc);
      acc[t] = wmma_bf16(pf.v, bv, acc[t]);
    }
  }
  const float g = gamma[0];
  for (int t = 0; t < 16; t++) {
    int d = t * 16 + nn;
    for (int r = 0; r < 8; r++) {
      int n = q0 + r + 8 * half;
      float o = acc[t][r] / lrow[r];
      float xv = x[((size_t)b * SEQ + n) * DIMD + d];
      xglbT[(size_t)b * DIMD * SEQ + (size_t)d * SEQ + n] = (bf16_t)(g * o + xv);
    }
  }
}

// ---------------- K3: dynamic_adj = sigmoid(fc_w @ concat(x_glb,x) + fc_b), stored transposed ----------------
__global__ __launch_bounds__(256) void k_fc(const bf16_t* __restrict__ fcw,
                                            const float* __restrict__ fcb,
                                            const bf16_t* __restrict__ xglbT,
                                            const bf16_t* __restrict__ xT,
                                            bf16_t* __restrict__ adjT) {
  const int b = blockIdx.y;
  const int wave = threadIdx.x >> 5;
  const int tile = blockIdx.x * 8 + wave;        // 8 x 8 tiles
  const int m0 = (tile / 8) * 32, n0 = (tile % 8) * 32;
  const bf16_t* Bg = xglbT + (size_t)b * DIMD * SEQ;
  const bf16_t* Bx = xT    + (size_t)b * DIMD * SEQ;
  v8f acc[2][2] = {};
  for (int k0 = 0; k0 < 2048; k0 += 32) {
    const bf16_t* Brow = (k0 < 1024) ? Bg : Bx;
    int kk = (k0 < 1024) ? k0 : k0 - 1024;
    v16bf a0 = load_a_frag(fcw + (size_t)m0 * 2048, 2048, k0);
    v16bf a1 = load_a_frag(fcw + (size_t)(m0 + 16) * 2048, 2048, k0);
    v16bf b0 = load_b_frag(Brow + (size_t)n0 * SEQ, SEQ, kk);
    v16bf b1 = load_b_frag(Brow + (size_t)(n0 + 16) * SEQ, SEQ, kk);
    acc[0][0] = wmma_bf16(a0, b0, acc[0][0]);
    acc[0][1] = wmma_bf16(a0, b1, acc[0][1]);
    acc[1][0] = wmma_bf16(a1, b0, acc[1][0]);
    acc[1][1] = wmma_bf16(a1, b1, acc[1][1]);
  }
  const int lane = threadIdx.x & 31, half = lane >> 4, nn = lane & 15;
  for (int i = 0; i < 2; i++)
    for (int j = 0; j < 2; j++)
      for (int r = 0; r < 8; r++) {
        int m = m0 + 16 * i + r + 8 * half;      // o index
        int n = n0 + 16 * j + nn;                // l index
        float v = acc[i][j][r] + fcb[m];
        v = 1.f / (1.f + __expf(-v));
        adjT[(size_t)b * DIMD * DIMD + (size_t)n * DIMD + m] = (bf16_t)v;
      }
}

// ---------------- K4: y = leaky(x @ adj), stored transposed ----------------
__global__ __launch_bounds__(256) void k_dyn(const bf16_t* __restrict__ xbf,
                                             const bf16_t* __restrict__ adjT,
                                             bf16_t* __restrict__ yT) {
  const int b = blockIdx.y;
  const int wave = threadIdx.x >> 5;
  const int tile = blockIdx.x * 8 + wave;        // 32 x 8 tiles
  const int m0 = (tile / 8) * 32, n0 = (tile % 8) * 32;
  const bf16_t* A = xbf + (size_t)b * SEQ * DIMD;
  const bf16_t* Bb = adjT + (size_t)b * DIMD * DIMD;
  v8f acc[2][2] = {};
  for (int k0 = 0; k0 < DIMD; k0 += 32) {
    v16bf a0 = load_a_frag(A + (size_t)m0 * DIMD, DIMD, k0);
    v16bf a1 = load_a_frag(A + (size_t)(m0 + 16) * DIMD, DIMD, k0);
    v16bf b0 = load_b_frag(Bb + (size_t)n0 * DIMD, DIMD, k0);
    v16bf b1 = load_b_frag(Bb + (size_t)(n0 + 16) * DIMD, DIMD, k0);
    acc[0][0] = wmma_bf16(a0, b0, acc[0][0]);
    acc[0][1] = wmma_bf16(a0, b1, acc[0][1]);
    acc[1][0] = wmma_bf16(a1, b0, acc[1][0]);
    acc[1][1] = wmma_bf16(a1, b1, acc[1][1]);
  }
  const int lane = threadIdx.x & 31, half = lane >> 4, nn = lane & 15;
  for (int i = 0; i < 2; i++)
    for (int j = 0; j < 2; j++)
      for (int r = 0; r < 8; r++) {
        int m = m0 + 16 * i + r + 8 * half;      // token index
        int n = n0 + 16 * j + nn;                // l index
        yT[(size_t)b * DIMD * SEQ + (size_t)n * SEQ + m] = (bf16_t)leaky(acc[i][j][r]);
      }
}

// ---------------- K5: out = leaky(dw_w @ y + dw_b), f32 output ----------------
__global__ __launch_bounds__(256) void k_out(const bf16_t* __restrict__ dww,
                                             const float* __restrict__ dwb,
                                             const bf16_t* __restrict__ yT,
                                             float* __restrict__ out) {
  const int b = blockIdx.y;
  const int wave = threadIdx.x >> 5;
  const int tile = blockIdx.x * 8 + wave;        // 32 x 8 tiles
  const int m0 = (tile / 8) * 32, n0 = (tile % 8) * 32;
  const bf16_t* Bb = yT + (size_t)b * DIMD * SEQ;
  v8f acc[2][2] = {};
  for (int k0 = 0; k0 < 1024; k0 += 32) {
    v16bf a0 = load_a_frag(dww + (size_t)m0 * 1024, 1024, k0);
    v16bf a1 = load_a_frag(dww + (size_t)(m0 + 16) * 1024, 1024, k0);
    v16bf b0 = load_b_frag(Bb + (size_t)n0 * SEQ, SEQ, k0);
    v16bf b1 = load_b_frag(Bb + (size_t)(n0 + 16) * SEQ, SEQ, k0);
    acc[0][0] = wmma_bf16(a0, b0, acc[0][0]);
    acc[0][1] = wmma_bf16(a0, b1, acc[0][1]);
    acc[1][0] = wmma_bf16(a1, b0, acc[1][0]);
    acc[1][1] = wmma_bf16(a1, b1, acc[1][1]);
  }
  const int lane = threadIdx.x & 31, half = lane >> 4, nn = lane & 15;
  for (int i = 0; i < 2; i++)
    for (int j = 0; j < 2; j++)
      for (int r = 0; r < 8; r++) {
        int m = m0 + 16 * i + r + 8 * half;      // o index
        int n = n0 + 16 * j + nn;                // l index
        out[((size_t)b * 1024 + m) * DIMD + n] = leaky(acc[i][j][r] + dwb[m]);
      }
}

extern "C" void kernel_launch(void* const* d_in, const int* in_sizes, int n_in,
                              void* d_out, int out_size, void* d_ws, size_t ws_size,
                              hipStream_t stream) {
  const float* x     = (const float*)d_in[0];
  const float* qkv_w = (const float*)d_in[1];
  const float* qkv_b = (const float*)d_in[2];
  const float* gamma = (const float*)d_in[3];
  const float* fc_w  = (const float*)d_in[4];
  const float* fc_b  = (const float*)d_in[5];
  const float* dw_w  = (const float*)d_in[6];
  const float* dw_b  = (const float*)d_in[7];
  float* out = (float*)d_out;
  char* ws = (char*)d_ws;
  const size_t MB = 1024 * 1024;

  bf16_t* xbf = (bf16_t*)(ws + 0);        // 32 MB
  bf16_t* xT  = (bf16_t*)(ws + 32 * MB);  // 32 MB
  bf16_t* qw  = (bf16_t*)(ws + 64 * MB);  // 0.375 MB
  bf16_t* fcw = (bf16_t*)(ws + 65 * MB);  // 1 MB
  bf16_t* dww = (bf16_t*)(ws + 66 * MB);  // 2 MB
  bf16_t* qb  = (bf16_t*)(ws + 68 * MB);  // 32 MB
  bf16_t* kb  = (bf16_t*)(ws + 100 * MB); // 32 MB
  bf16_t* vT  = (bf16_t*)(ws + 132 * MB); // 32 MB
  bf16_t* xgT = (bf16_t*)(ws + 164 * MB); // 32 MB  (total 196 MB)
  bf16_t* adjT = qb;                      // reuse: q dead after attention
  bf16_t* yT   = kb;                      // reuse: k dead after attention

  k_cvt<<<4096, 256, 0, stream>>>(x, xbf, BATCH * SEQ * DIMD);
  k_cvt<<<768, 256, 0, stream>>>(qkv_w, qw, QKV3 * DIMD);
  k_cvt<<<2048, 256, 0, stream>>>(fc_w, fcw, DIMD * 2048);
  k_cvt<<<4096, 256, 0, stream>>>(dw_w, dww, 1024 * 1024);
  k_xT<<<(BATCH * SEQ * DIMD + 255) / 256, 256, 0, stream>>>(x, xT);

  k_qkv<<<dim3(96, BATCH), 256, 0, stream>>>(xbf, qw, qkv_b, qb, kb, vT);
  k_attn<<<dim3(8, BATCH), 256, 0, stream>>>(qb, kb, vT, x, gamma, xgT);
  k_fc<<<dim3(8, BATCH), 256, 0, stream>>>(fcw, fc_b, xgT, xT, adjT);
  k_dyn<<<dim3(32, BATCH), 256, 0, stream>>>(xbf, adjT, yT);
  k_out<<<dim3(32, BATCH), 256, 0, stream>>>(dww, dw_b, yT, out);
}
